// WidthCharTransformerMLPHeaderFastPosV2_63754494542661
// MI455X (gfx1250) — matrix operand order repaired and checked by reading, not compiled, not run on hardware
//
#include <hip/hip_runtime.h>
#include <hip/hip_bf16.h>

typedef __attribute__((ext_vector_type(16))) _Float16 v16h;
typedef __attribute__((ext_vector_type(8)))  _Float16 v8h;
typedef __attribute__((ext_vector_type(4)))  _Float16 v4h;
typedef __attribute__((ext_vector_type(8)))  float    v8f;

// Problem constants: N=2048, L=32, V=8192, H=8, d=128
#define QH_STRIDE 136   // halves: 272B row, 16B multiple
#define KT_STRIDE 40    // halves: 80B row, 16B multiple
#define PH_STRIDE 40
#define W_STRIDE  40
#define S_STRIDE  33    // floats
#define OB_STRIDE 132   // floats

struct __align__(16) SharedBlk {
    // f16 arrays first; every size is a multiple of 16B so casts stay aligned
    _Float16 qh[32 * QH_STRIDE];     // Q rows (f16) [l][d]
    _Float16 vh[32 * QH_STRIDE];     // V rows (f16) [key][d]
    _Float16 kt[128 * KT_STRIDE];    // K transposed (f16) [d][key]
    _Float16 Ph[32 * PH_STRIDE];     // softmax probs (f16) [q][key]
    _Float16 ln2[32 * 128 * 8];      // LN'd out (f16) [l][c][h] (h innermost)
    _Float16 W1p[32 * W_STRIDE];     // W1 zero-padded to 32x32 (rows >=8 are 0)
    _Float16 W2p[32 * W_STRIDE];     // W2 zero-padded to 32x16 (cols >=8 are 0)
    _Float16 hs[4 * 16 * W_STRIDE];  // per-wave relu(h1) scratch, 16x32 used
    float    S[32 * S_STRIDE];       // scores f32
    float    obuf[32 * OB_STRIDE];   // attention out f32 [q][d]
    float    pos[32 * 128];          // positional encodings [l][c]
    float    b1s[32];
    float    b2s[8];
    float    gs[128];
    float    bs[128];
    float    mur[32];                // per-row mean
    float    rsr[32];                // per-row rstd
    float    rowred[256];            // LN partials [row][q4]{sum,sumsq}
    int      toks[32];
    float    msumf;
};

__device__ __forceinline__ v16h cat8(v8h a, v8h b) {
    return __builtin_shufflevector(a, b, 0,1,2,3,4,5,6,7,8,9,10,11,12,13,14,15);
}

__global__ __launch_bounds__(128)
void attn_fused_kernel(const int* __restrict__ inputs, const float* __restrict__ qkv,
                       const float* __restrict__ gamma, const float* __restrict__ beta,
                       const float* __restrict__ W1, const float* __restrict__ b1,
                       const float* __restrict__ W2, const float* __restrict__ b2,
                       float* __restrict__ word_ft)
{
    __shared__ SharedBlk sm;
    const int tid  = threadIdx.x;       // 0..127
    const int n    = blockIdx.x;        // sequence
    const int lane = tid & 31;
    const int w    = tid >> 5;          // wave 0..3
    const int M    = lane & 15;
    const int hf   = lane >> 4;         // half-wave 0/1

    // ---- small weights (layer weights as zero-padded f16 WMMA B-operands) ----
    for (int idx = tid; idx < 32 * 32; idx += 128) {
        int k = idx >> 5, j = idx & 31;
        sm.W1p[k * W_STRIDE + j] = (k < 8) ? (_Float16)W1[k * 32 + j] : (_Float16)0.f;
    }
    for (int idx = tid; idx < 32 * 16; idx += 128) {
        int k = idx >> 4, j = idx & 15;
        sm.W2p[k * W_STRIDE + j] = (j < 8) ? (_Float16)W2[k * 8 + j] : (_Float16)0.f;
    }
    if (tid < 32) sm.b1s[tid] = b1[tid];
    if (tid < 8)  sm.b2s[tid] = b2[tid];
    sm.gs[tid] = gamma[tid];
    sm.bs[tid] = beta[tid];
    if (tid < 32) sm.toks[tid] = inputs[n * 32 + tid];
    {   // positional table: pos[l][c] = c<64 ? cos(l*inv(c)) : sin(l*inv(c-64))
        int c = tid;
        int j = (c < 64) ? c : (c - 64);
        float invc = expf(-(float)j * (9.210340371976184f / 64.0f)); // 10000^(-j/64)
        for (int l = 0; l < 32; ++l) {
            float ph = (float)l * invc;
            sm.pos[l * 128 + c] = (c < 64) ? cosf(ph) : sinf(ph);
        }
    }
    __syncthreads();
    if (tid == 0) {
        int s = 0;
        for (int l = 0; l < 32; ++l) s += (sm.toks[l] != 0) ? 1 : 0;
        sm.msumf = (float)s;
    }

    const float scale = 0.08838834764831845f;  // 1/sqrt(128)

    for (int h = 0; h < 8; ++h) {
        // ---- gather qkv rows + pos -> f16 LDS; one row per wave, b128 loads ----
        // 96 rows = (l,j); rowid = it*4 + w keeps j wave-uniform.
        for (int it = 0; it < 24; ++it) {
            int rowid = it * 4 + w;          // 0..95
            int l = rowid / 3;
            int j = rowid - l * 3;
            int t = sm.toks[l];
            int c4 = lane * 4;               // 0..124
            const float4 g4 = *(const float4*)(&qkv[(((size_t)t * 3 + j) * 8 + h) * 128 + c4]);
            const float4 p4 = *(const float4*)(&sm.pos[l * 128 + c4]);
            v4h hv;
            hv[0] = (_Float16)(g4.x + p4.x);
            hv[1] = (_Float16)(g4.y + p4.y);
            hv[2] = (_Float16)(g4.z + p4.z);
            hv[3] = (_Float16)(g4.w + p4.w);
            if (j == 0)      *(v4h*)(&sm.qh[l * QH_STRIDE + c4]) = hv;
            else if (j == 2) *(v4h*)(&sm.vh[l * QH_STRIDE + c4]) = hv;
            else {
                #pragma unroll
                for (int i = 0; i < 4; ++i) sm.kt[(c4 + i) * KT_STRIDE + l] = hv[i];
            }
        }
        __syncthreads();

        // ---- S = Q K^T / sqrt(d): 4 tiles of 16x16, one per wave ----
        {
            int tm = w >> 1, tn = w & 1;
            v8f acc = {};
            #pragma unroll
            for (int kb = 0; kb < 128; kb += 32) {
                const v8h* qa = (const v8h*)(&sm.qh[(tm * 16 + M) * QH_STRIDE + kb + 8 * hf]);
                v16h a = cat8(qa[0], qa[2]);       // K = kb+8hf..+7, kb+16+8hf..+7
                v16h b;
                #pragma unroll
                for (int s = 0; s < 16; ++s)
                    b[s] = sm.kt[(kb + (s < 8 ? 8 * hf + s : 8 + 8 * hf + s)) * KT_STRIDE + tn * 16 + M];
                acc = __builtin_amdgcn_wmma_f32_16x16x32_f16(
                        false, a, false, b, (short)0, acc, false, false);
            }
            int col = tn * 16 + M;
            bool masked = (sm.toks[col] == 0);
            #pragma unroll
            for (int r = 0; r < 8; ++r)
                sm.S[(tm * 16 + r + 8 * hf) * S_STRIDE + col] =
                    masked ? -1.0e9f : acc[r] * scale;
        }
        __syncthreads();

        // ---- softmax rows -> Ph (f16) ----
        if (tid < 32) {
            int r = tid;
            float m = -3.0e38f;
            for (int k2 = 0; k2 < 32; ++k2) m = fmaxf(m, sm.S[r * S_STRIDE + k2]);
            float e[32];
            float s = 0.f;
            for (int k2 = 0; k2 < 32; ++k2) {
                float ev = expf(sm.S[r * S_STRIDE + k2] - m);
                e[k2] = ev; s += ev;
            }
            float inv = 1.0f / s;
            for (int k2 = 0; k2 < 32; ++k2)
                sm.Ph[r * PH_STRIDE + k2] = (_Float16)(e[k2] * inv);
        }
        __syncthreads();

        // ---- O = P V : 16 tiles (2x8), 4 per wave; A hoisted (tm fixed per wave) ----
        {
            int tm = w >> 1;
            const v8h* pa = (const v8h*)(&sm.Ph[(tm * 16 + M) * PH_STRIDE + 8 * hf]);
            v16h a = cat8(pa[0], pa[2]);           // K(keys) 8hf..+7, 16+8hf..+7
            int key = M + 16 * hf;                 // B row = key
            #pragma unroll
            for (int ti = 0; ti < 4; ++ti) {
                int tn = (w & 1) * 4 + ti;
                v16h b;
                #pragma unroll
                for (int s = 0; s < 16; ++s)
                    b[s] = sm.vh[(s < 8 ? 8 * hf + s : 8 + 8 * hf + s) * QH_STRIDE + tn * 16 + M];
                v8f acc = {};
                acc = __builtin_amdgcn_wmma_f32_16x16x32_f16(
                        false, a, false, b, (short)0, acc, false, false);
                #pragma unroll
                for (int r = 0; r < 8; ++r)
                    sm.obuf[(tm * 16 + r + 8 * hf) * OB_STRIDE + tn * 16 + M] = acc[r];
            }
        }
        __syncthreads();

        // ---- LayerNorm over d=128 ----
        {   // partial sums: 4 threads per row
            int r = tid >> 2, q4 = tid & 3;
            float s = 0.f, s2 = 0.f;
            for (int c = q4 * 32; c < q4 * 32 + 32; ++c) {
                float x = sm.obuf[r * OB_STRIDE + c];
                s += x; s2 += x * x;
            }
            sm.rowred[(r * 4 + q4) * 2 + 0] = s;
            sm.rowred[(r * 4 + q4) * 2 + 1] = s2;
        }
        __syncthreads();
        if (tid < 32) {
            int r = tid;
            float s = 0.f, s2 = 0.f;
            #pragma unroll
            for (int i = 0; i < 4; ++i) {
                s  += sm.rowred[(r * 4 + i) * 2 + 0];
                s2 += sm.rowred[(r * 4 + i) * 2 + 1];
            }
            float mu  = s * (1.0f / 128.0f);
            float var = s2 * (1.0f / 128.0f) - mu * mu;
            sm.mur[r] = mu;
            sm.rsr[r] = rsqrtf(var + 1e-5f);
        }
        __syncthreads();
        {   // normalize, c-parallel; write ln2[l][c][h] (h innermost for WMMA A rows)
            int c = tid;
            float gg = sm.gs[c], bb = sm.bs[c];
            for (int r = 0; r < 32; ++r) {
                float y = (sm.obuf[r * OB_STRIDE + c] - sm.mur[r]) * sm.rsr[r] * gg + bb;
                sm.ln2[(r * 128 + c) * 8 + h] = (_Float16)y;
            }
        }
        __syncthreads();
    } // heads

    // ---- MLP via WMMA + masked mean over L, accumulated in the C operand ----
    // rows p = l*128 + c; a 16-row tile = 16 consecutive c at fixed l.
    // layer1: relu(X(16x8, K zero-padded to 32) @ W1p + b1)  -> 2 WMMAs (N tiles)
    // layer2: h1(16x32) @ W2p, C accumulates sum over masked l -> 1 WMMA
    {
        const int n0 = M;                       // output column (j), valid for n0<8
        const int krow = M + 16 * hf;           // B-operand row for this lane
        _Float16* hsw = &sm.hs[w * 16 * W_STRIDE];
        float invm = 1.0f / sm.msumf;

        v16h B1a, B1b, B2;
        #pragma unroll
        for (int s = 0; s < 16; ++s) {
            int kr = (s < 8) ? 8 * hf + s : 8 + 8 * hf + s;
            B1a[s] = sm.W1p[kr * W_STRIDE + n0];
            B1b[s] = sm.W1p[kr * W_STRIDE + 16 + n0];
            B2[s]  = sm.W2p[kr * W_STRIDE + n0];
        }
        v8f c1a, c1b;
        {
            float bva = sm.b1s[n0], bvb = sm.b1s[16 + n0];
            #pragma unroll
            for (int r = 0; r < 8; ++r) { c1a[r] = bva; c1b[r] = bvb; }
        }

        #pragma unroll
        for (int cgi = 0; cgi < 2; ++cgi) {
            int cg = w + cgi * 4;               // c-group 0..7 (16 channels each)
            v8f acc2 = {};
            for (int l = 0; l < 32; ++l) {
                _Float16 mf = (sm.toks[l] != 0) ? (_Float16)1.f : (_Float16)0.f;
                // A1: rows = c in [cg*16, cg*16+16); halves K = heads (zero-padded)
                const v8h* xp = (const v8h*)(&sm.ln2[(l * 128 + cg * 16 + M) * 8]);
                v8h x8 = xp[0];
                v16h a1;
                #pragma unroll
                for (int i = 0; i < 8; ++i) {
                    a1[i]     = hf ? (_Float16)0.f : x8[i];
                    a1[i + 8] = (_Float16)0.f;
                }
                v8f h1a = __builtin_amdgcn_wmma_f32_16x16x32_f16(
                            false, a1, false, B1a, (short)0, c1a, false, false);
                v8f h1b = __builtin_amdgcn_wmma_f32_16x16x32_f16(
                            false, a1, false, B1b, (short)0, c1b, false, false);
                // relu -> f16 -> wave-private LDS scratch (layout [m][32])
                #pragma unroll
                for (int r = 0; r < 8; ++r) {
                    int m = r + 8 * hf;
                    hsw[m * W_STRIDE + n0]      = (_Float16)fmaxf(h1a[r], 0.f);
                    hsw[m * W_STRIDE + 16 + n0] = (_Float16)fmaxf(h1b[r], 0.f);
                }
                // A2 = relu(h1), masked (branchless, EXEC stays all-ones)
                const v8h* ap = (const v8h*)(&hsw[M * W_STRIDE + 8 * hf]);
                v16h a2 = cat8(ap[0], ap[2]);
                #pragma unroll
                for (int i = 0; i < 16; ++i) a2[i] *= mf;
                acc2 = __builtin_amdgcn_wmma_f32_16x16x32_f16(
                        false, a2, false, B2, (short)0, acc2, false, false);
            }
            // word_ft[n][c*8+j] = acc2/msum + b2[j]
            if (n0 < 8) {
                float bb = sm.b2s[n0];
                #pragma unroll
                for (int r = 0; r < 8; ++r) {
                    int c = cg * 16 + r + 8 * hf;
                    word_ft[(size_t)n * 1024 + c * 8 + n0] = acc2[r] * invm + bb;
                }
            }
        }
    }
}

// name_ft: mean over groups of 8 sequences -> (256,1024) == (64,4,1024) flat
__global__ __launch_bounds__(256)
void name_reduce_kernel(const float* __restrict__ word_ft, float* __restrict__ out)
{
    int g = blockIdx.x;                          // 0..255
    for (int f = threadIdx.x; f < 1024; f += 256) {
        float s = 0.f;
        #pragma unroll
        for (int w = 0; w < 8; ++w)
            s += word_ft[((size_t)g * 8 + w) * 1024 + f];
        out[(size_t)g * 1024 + f] = s * 0.125f;
    }
}

extern "C" void kernel_launch(void* const* d_in, const int* in_sizes, int n_in,
                              void* d_out, int out_size, void* d_ws, size_t ws_size,
                              hipStream_t stream)
{
    (void)in_sizes; (void)n_in; (void)out_size; (void)ws_size;
    const int*   inputs = (const int*)  d_in[0];
    const float* qkv    = (const float*)d_in[1];
    const float* gamma  = (const float*)d_in[2];
    const float* beta   = (const float*)d_in[3];
    const float* W1     = (const float*)d_in[4];
    const float* b1     = (const float*)d_in[5];
    const float* W2     = (const float*)d_in[6];
    const float* b2     = (const float*)d_in[7];

    float* word = (float*)d_ws;      // 2048*1024 f32 = 8 MB scratch
    float* out  = (float*)d_out;     // 256*1024 f32

    attn_fused_kernel<<<2048, 128, 0, stream>>>(inputs, qkv, gamma, beta,
                                                W1, b1, W2, b2, word);
    name_reduce_kernel<<<256, 256, 0, stream>>>(word, out);
}
